// EthanolSNN_51049981280307
// MI455X (gfx1250) — matrix operand order repaired and checked by reading, not compiled, not run on hardware
//
#include <hip/hip_runtime.h>

typedef __attribute__((ext_vector_type(16))) _Float16 v16h;
typedef __attribute__((ext_vector_type(8)))  _Float16 h8;
typedef __attribute__((ext_vector_type(8)))  float    v8f;
typedef __attribute__((ext_vector_type(2)))  float    f32x2;

#define T_STEPS 2048
#define BATCH   2048
#define D_IN    30
#define H1      28
#define H2      14
#define BETA    0.9f
#define THR     1.0f
#define NBLK    (T_STEPS / 16)

static __device__ __forceinline__ v8f wmma_f16(v16h a, v16h b, v8f c) {
  // D = A(16x32 f16) x B(32x16 f16) + C(16x16 f32)
  return __builtin_amdgcn_wmma_f32_16x16x32_f16(false, a, false, b, (short)0, c, false, false);
}

static __device__ __forceinline__ float bcast_lane(float v, int lane) {
  return __uint_as_float(__builtin_amdgcn_readlane(__float_as_uint(v), lane));
}

// Swap the two 16-lane halves of the wave (lane i <-> lane i^16) on the VALU
// (v_permlanex16_b32 with identity lane selects) -- no DS traffic, no dscnt waits.
static __device__ __forceinline__ float xor16(float v) {
  int i = __float_as_int(v);
  int r = __builtin_amdgcn_permlanex16(i, i, 0x76543210, (int)0xfedcba98, false, false);
  return __int_as_float(r);
}

// Gather one wave's A-operand (16 timesteps x 32 K) from a staged 16x32 f16 LDS tile.
// A layout (16-bit, 16x32): lane L<16 holds M=L, K {0..7,16..23}; lane L>=16 holds M=L-16, K {8..15,24..31}.
static __device__ __forceinline__ v16h pack_lds_tile(const _Float16* row, bool lo) {
  const h8* r = (const h8*)(row + (lo ? 0 : 8));
  h8 q0 = r[0];   // halfs base+0..7   (one ds_load_b128)
  h8 q1 = r[2];   // halfs base+16..23 (one ds_load_b128)
  v16h a;
  #pragma unroll
  for (int e = 0; e < 8; ++e) { a[e] = q0[e]; a[8 + e] = q1[e]; }
  return a;
}

struct XRegs { f32x2 p0, p1, p2, p3, p4, p5, p6, p7; };

static __device__ __forceinline__ XRegs load_x(const float* px, bool lo) {
  XRegs r;
  r.p0 = *(const f32x2*)(px + 0);
  r.p1 = *(const f32x2*)(px + 2);
  r.p2 = *(const f32x2*)(px + 4);
  r.p3 = *(const f32x2*)(px + 6);
  r.p4 = *(const f32x2*)(px + 16);
  r.p5 = *(const f32x2*)(px + 18);
  r.p6 = *(const f32x2*)(px + 20);
  r.p7 = *(const f32x2*)(px + (lo ? 22 : 20)); // hi: harmless dup of cols 28,29 (avoids OOB)
  return r;
}

static __device__ __forceinline__ v16h make_A(const XRegs& r, bool lo) {
  v16h A;
  A[0]=(_Float16)r.p0.x;  A[1]=(_Float16)r.p0.y;
  A[2]=(_Float16)r.p1.x;  A[3]=(_Float16)r.p1.y;
  A[4]=(_Float16)r.p2.x;  A[5]=(_Float16)r.p2.y;
  A[6]=(_Float16)r.p3.x;  A[7]=(_Float16)r.p3.y;
  A[8]=(_Float16)r.p4.x;  A[9]=(_Float16)r.p4.y;
  A[10]=(_Float16)r.p5.x; A[11]=(_Float16)r.p5.y;
  A[12]=(_Float16)r.p6.x; A[13]=(_Float16)r.p6.y;
  A[14]= lo ? (_Float16)r.p7.x : (_Float16)1.0f;   // hi lanes: K=30 is the bias column (1.0)
  A[15]= lo ? (_Float16)r.p7.y : (_Float16)0.0f;   // hi lanes: K=31 zero pad
  return A;
}

__global__ __launch_bounds__(256)
void snn_scan_kernel(const float* __restrict__ x,
                     const float* __restrict__ W1, const float* __restrict__ b1,
                     const float* __restrict__ W2, const float* __restrict__ b2,
                     const float* __restrict__ W3, const float* __restrict__ b3,
                     float* __restrict__ out)
{
  // Per-wave private spike staging tiles (16 timesteps x 32 cols, f16), 8 waves per block.
  __shared__ __align__(16) _Float16 s_spk[8][2][16][32];

  const int tid   = threadIdx.x;
  const int lane  = tid & 31;
  const int wid   = tid >> 5;
  const int bidx  = blockIdx.x * 8 + wid;   // batch element, one wave each
  const bool lo   = (lane < 16);
  const int  mrow = lane & 15;
  const int  khalf = lo ? 0 : 16;

  // ---------------- Constant B operands (f16), bias folded in as K row K_in ----------------
  // B layout (16-bit, 32x16): lane L<16: N=L, K {0..7,16..23}; lane L>=16: N=L-16, K {8..15,24..31}.
  v16h B1a, B1b, B2t, B3t;
  #pragma unroll
  for (int e = 0; e < 16; ++e) {
    const int k = (e < 8) ? (khalf >> 1) + e : 16 + (khalf >> 1) + (e - 8);
    const int n = mrow;
    float v;
    // layer 1, neurons 0..15
    v = 0.f;
    if (k < D_IN) v = W1[n * D_IN + k]; else if (k == D_IN) v = b1[n];
    B1a[e] = (_Float16)v;
    // layer 1, neurons 16..27 (28..31 zero-padded)
    const int n2 = 16 + n;
    v = 0.f;
    if (n2 < H1) { if (k < D_IN) v = W1[n2 * D_IN + k]; else if (k == D_IN) v = b1[n2]; }
    B1b[e] = (_Float16)v;
    // layer 2: 14 neurons, K = 28 spikes + bias row at K=28, rows 29..31 zero
    v = 0.f;
    if (n < H2) { if (k < H1) v = W2[n * H1 + k]; else if (k == H1) v = b2[n]; }
    B2t[e] = (_Float16)v;
    // layer 3: 1 neuron, K = 14 spikes + bias row at K=14, rest zero
    v = 0.f;
    if (n == 0) { if (k < H2) v = W3[k]; else if (k == H2) v = b3[0]; }
    B3t[e] = (_Float16)v;
  }

  _Float16 (*s1)[32] = s_spk[wid][0];
  _Float16 (*s2)[32] = s_spk[wid][1];

  // Per-lane x pointer: row (t0+mrow) of batch bidx, column base 0 (lo) / 8 (hi). 8B aligned.
  const float* px = x + ((size_t)mrow * BATCH + bidx) * D_IN + (lo ? 0 : 8);
  const unsigned outbase = (unsigned)(lane * BATCH + bidx);

  float m1 = 0.f, m2 = 0.f, m3 = 0.f;

  // Software pipeline: x registers for the current block are loaded one iteration ahead.
  XRegs xr = load_x(px, lo);

  for (int blk = 0; blk < NBLK; ++blk) {
    // ---------------- Layer-1 A tile: x for 16 timesteps, f32 -> f16 ----------------
    v16h A = make_A(xr, lo);

    // Prefetch next block's x (last iteration re-reads current block: in-bounds, discarded).
    const float* pn = (blk + 1 < NBLK) ? (px + 16 * BATCH * D_IN) : px;
    px = pn;
    xr = load_x(px, lo);

    v8f zero = {};
    v8f d0 = wmma_f16(A, B1a, zero);   // cur1, neurons 0..15, 16 timesteps
    v8f d1 = wmma_f16(A, B1b, zero);   // cur1, neurons 16..27

    // ---------------- Layer-1 membrane scan; spikes -> LDS (transpose staging) ----------------
    #pragma unroll
    for (int M = 0; M < 16; ++M) {
      const float a0 = (M < 8) ? d0[M] : d0[M - 8];
      const float a1 = (M < 8) ? d1[M] : d1[M - 8];
      float cur;
      if (M < 8) { const float t = xor16(a1); cur = lo ? a0 : t; }
      else       { const float t = xor16(a0); cur = lo ? t  : a1; }
      const float rst = (m1 > THR) ? THR : 0.f;     // heaviside(m_prev - thr) * thr
      m1 = BETA * m1 + cur - rst;
      const _Float16 spk = (m1 > THR) ? (_Float16)1.0f : (_Float16)0.0f;
      s1[M][lane] = (lane == 28) ? (_Float16)1.0f : spk;  // col 28 = bias column for layer 2
    }
    __asm__ __volatile__("" ::: "memory");          // keep DS loads after DS stores (HW is in-order)

    // ---------------- Layer 2: spikes(16x28+bias) @ W2^T via one WMMA ----------------
    v16h A2 = pack_lds_tile(&s1[mrow][0], lo);
    v8f d2 = wmma_f16(A2, B2t, zero);

    #pragma unroll
    for (int M = 0; M < 16; ++M) {
      const float cur = (M < 8) ? d2[M] : xor16(d2[M - 8]);
      const float rst = (m2 > THR) ? THR : 0.f;
      m2 = BETA * m2 + cur - rst;
      const _Float16 spk = (m2 > THR) ? (_Float16)1.0f : (_Float16)0.0f;
      s2[M][lane] = (lane == 14) ? (_Float16)1.0f : spk;  // col 14 = bias column for layer 3
    }
    __asm__ __volatile__("" ::: "memory");

    // ---------------- Layer 3: spikes(16x14+bias) @ W3^T via one WMMA ----------------
    v16h A3 = pack_lds_tile(&s2[mrow][0], lo);
    v8f d3 = wmma_f16(A3, B3t, zero);

    // ---------------- m3 scan (wave-uniform), gather 16 outputs into lanes 0..15 ----------------
    float out_val = 0.f;
    #pragma unroll
    for (int M = 0; M < 16; ++M) {
      const float c3 = (M < 8) ? bcast_lane(d3[M], 0) : bcast_lane(d3[M - 8], 16);
      const float rst = (m3 > THR) ? THR : 0.f;
      m3 = BETA * m3 + c3 - rst;
      out_val = (lane == M) ? m3 : out_val;
    }
    if (lo) out[outbase + (unsigned)blk * (16u * BATCH)] = out_val;  // out[t0+lane, bidx]
  }
}

extern "C" void kernel_launch(void* const* d_in, const int* in_sizes, int n_in,
                              void* d_out, int out_size, void* d_ws, size_t ws_size,
                              hipStream_t stream) {
  const float* x  = (const float*)d_in[0];
  const float* W1 = (const float*)d_in[1];
  const float* b1 = (const float*)d_in[2];
  const float* W2 = (const float*)d_in[3];
  const float* b2 = (const float*)d_in[4];
  const float* W3 = (const float*)d_in[5];
  const float* b3 = (const float*)d_in[6];
  float* out = (float*)d_out;

  dim3 block(256);                 // 8 waves (wave32) per workgroup
  dim3 grid(BATCH / 8);            // one wave per batch element -> 2048 waves
  snn_scan_kernel<<<grid, block, 0, stream>>>(x, W1, b1, W2, b2, W3, b3, out);
}
